// Attention_70085276336431
// MI455X (gfx1250) — matrix operand order repaired and checked by reading, not compiled, not run on hardware
//
#include <hip/hip_runtime.h>
#include <hip/hip_bf16.h>
#include <math.h>

// Problem constants (from reference): B=64, S=2048, H2=512
#define BATCH 64
#define SEQ   2048
#define HD    512

typedef __attribute__((ext_vector_type(16))) __bf16   v16bf;
typedef __attribute__((ext_vector_type(8)))  float    v8f;
typedef __attribute__((ext_vector_type(8)))  unsigned v8u;

__device__ __forceinline__ unsigned short f2bf(float f) {
    unsigned u = __float_as_uint(f);
    unsigned r = u + 0x7FFFu + ((u >> 16) & 1u);   // round-to-nearest-even
    return (unsigned short)(r >> 16);
}

// ---------------------------------------------------------------------------
// Prep: convert squish_w (512x512 f32) ONCE to bf16 in WMMA B-fragment order.
//   Bf[((nc*16 + kc)*32 + lane)*8 + v] = pack(W[k][n], W[k+1][n])
//   n = nc*16 + (lane&15),  k = kc*32 + (lane>>4)*16 + 2v
// Main kernel then reads B fragments with coalesced global_load_b128 (L2-hot).
// ---------------------------------------------------------------------------
__global__ __launch_bounds__(256) void
bprep_kernel(const float* __restrict__ squish_w, unsigned* __restrict__ Bf)
{
    int idx  = blockIdx.x * 256 + threadIdx.x;   // 0 .. 131071
    int v    = idx & 7;
    int lane = (idx >> 3) & 31;
    int kc   = (idx >> 8) & 15;
    int nc   = idx >> 12;
    int n    = nc * 16 + (lane & 15);
    int k    = kc * 32 + ((lane >> 4) << 4) + (v << 1);
    unsigned short lo = f2bf(squish_w[(long)k * HD + n]);
    unsigned short hi = f2bf(squish_w[(long)(k + 1) * HD + n]);
    Bf[idx] = (unsigned)lo | ((unsigned)hi << 16);
}

// ---------------------------------------------------------------------------
// Kernel 1: fused  scores[m] = sum_n tanh( (weight @ W)[m,n] ) * v[n]
// 8 waves/WG, 16 rows/wave. A slab (16x512 bf16) preloaded into 128 VGPRs via
// a 32-row LDS staging buffer (cycled 4x). Hot loop: global b128 B-fragment
// loads + v_wmma_f32_16x16x32_bf16 only — no LDS, no barriers, no converts.
// ---------------------------------------------------------------------------
#define ROWS_WG 128
#define SA 520   // padded bf16 row stride (1040 B) -> bank-conflict-free gathers

__global__ __launch_bounds__(256) void
att_scores_kernel(const float* __restrict__ weight,
                  const unsigned* __restrict__ Bf,
                  const float* __restrict__ atten_proj,
                  float* __restrict__ scores)
{
    __shared__ __align__(16) unsigned short A_lds[32 * SA];   // 33,280 B

    const int t    = threadIdx.x;
    const int wave = t >> 5;
    const int lane = t & 31;
    const int half = lane >> 4;
    const int l16  = lane & 15;
    const long rowBase = (long)blockIdx.x * ROWS_WG;

    // ---- stage A in 4 chunks of 32 rows; each wave grabs its fragments ----
    v16bf a[16];                                  // 128 VGPRs: 16 x (16x32 bf16)
    for (int ch = 0; ch < 4; ++ch) {
        for (int i = t; i < 32 * (HD / 4); i += 256) {
            int r  = i / (HD / 4);
            int c4 = (i % (HD / 4)) * 4;
            float4 vq = *(const float4*)(weight + (rowBase + ch * 32 + r) * HD + c4);
            unsigned short* dst = &A_lds[r * SA + c4];
            dst[0] = f2bf(vq.x); dst[1] = f2bf(vq.y);
            dst[2] = f2bf(vq.z); dst[3] = f2bf(vq.w);
        }
        __syncthreads();
        if ((wave >> 1) == ch) {                  // waves 2ch, 2ch+1 own this chunk
            int mloc = (wave & 1) * 16 + l16;
#pragma unroll
            for (int kc = 0; kc < 16; ++kc) {
                v8u au;
#pragma unroll
                for (int v = 0; v < 8; ++v) {
                    // A frag: contiguous k = 16*half + 2v (matches Bf packing)
                    int ka = kc * 32 + (half << 4) + (v << 1);
                    au[v] = *(const unsigned*)&A_lds[mloc * SA + ka];
                }
                a[kc] = __builtin_bit_cast(v16bf, au);
            }
        }
        __syncthreads();
    }

    float sacc[8];
#pragma unroll
    for (int i = 0; i < 8; ++i) sacc[i] = 0.0f;

    const v8u* __restrict__ bfrag = (const v8u*)Bf;   // [ (nc*16+kc)*32 + lane ]

    for (int nc = 0; nc < HD / 16; ++nc) {
        // prefetch next panel's first line for this lane
        if (nc + 1 < HD / 16)
            __builtin_prefetch(&bfrag[((nc + 1) * 16) * 32 + lane], 0, 3);

        v8f c = {};
#pragma unroll
        for (int kc = 0; kc < 16; ++kc) {
            v16bf b = __builtin_bit_cast(v16bf, bfrag[(nc * 16 + kc) * 32 + lane]);
            c = __builtin_amdgcn_wmma_f32_16x16x32_bf16(
                    false, a[kc], false, b, (short)0, c, false, false);
        }

        const float vproj = atten_proj[nc * 16 + l16];
        // C tile: VGPR i = row (i + 8*half), col = l16
#pragma unroll
        for (int i = 0; i < 8; ++i) sacc[i] += tanhf(c[i]) * vproj;
    }

    // reduce over the 16 lanes sharing each row-half
#pragma unroll
    for (int i = 0; i < 8; ++i) {
        float s = sacc[i];
        for (int off = 1; off < 16; off <<= 1) s += __shfl_xor(s, off, 16);
        sacc[i] = s;
    }
    if (l16 == 0) {
        long row = rowBase + wave * 16 + half * 8;
#pragma unroll
        for (int i = 0; i < 8; ++i) scores[row + i] = sacc[i];
    }
}

// ---------------------------------------------------------------------------
// Kernel 2: masked softmax over S per batch (in place in workspace).
// ---------------------------------------------------------------------------
__global__ __launch_bounds__(256) void
att_softmax_kernel(const float* __restrict__ mask, float* __restrict__ scores)
{
    __shared__ float red[256];
    const int b = blockIdx.x, t = threadIdx.x;
    float* sc = scores + (long)b * SEQ;
    const float* mk = mask + (long)b * SEQ;

    float vals[8];
    float mx = -INFINITY;
#pragma unroll
    for (int i = 0; i < 8; ++i) { vals[i] = sc[t * 8 + i]; mx = fmaxf(mx, vals[i]); }
    red[t] = mx; __syncthreads();
    for (int s2 = 128; s2 > 0; s2 >>= 1) {
        if (t < s2) red[t] = fmaxf(red[t], red[t + s2]);
        __syncthreads();
    }
    mx = red[0]; __syncthreads();

    float e[8], sum = 0.0f;
#pragma unroll
    for (int i = 0; i < 8; ++i) { e[i] = __expf(vals[i] - mx) * mk[t * 8 + i]; sum += e[i]; }
    red[t] = sum; __syncthreads();
    for (int s2 = 128; s2 > 0; s2 >>= 1) {
        if (t < s2) red[t] += red[t + s2];
        __syncthreads();
    }
    float inv = 1.0f / (red[0] + 1e-12f);
#pragma unroll
    for (int i = 0; i < 8; ++i) sc[t * 8 + i] = e[i] * inv;
}

// ---------------------------------------------------------------------------
// Kernel 3: out[b,d] = sum_s att[b,s] * x[b,s,d]   (pure HBM-bandwidth pass)
// ---------------------------------------------------------------------------
#define SCHUNKS 16
__global__ __launch_bounds__(256) void
att_pool_kernel(const float* __restrict__ x, const float* __restrict__ att,
                float* __restrict__ out)
{
    const int b = blockIdx.y, c = blockIdx.x, t = threadIdx.x;
    const int s0 = c * (SEQ / SCHUNKS);
    const float* xb = x + (long)b * SEQ * HD;
    const int d = t * 2;
    float ax = 0.0f, ay = 0.0f;
    for (int s = s0; s < s0 + SEQ / SCHUNKS; ++s) {
        if (s + 8 < SEQ)
            __builtin_prefetch(xb + (long)(s + 8) * HD + d, 0, 3);
        float a = att[(long)b * SEQ + s];               // uniform -> scalar path
        float2 xv = *(const float2*)(xb + (long)s * HD + d);
        ax += a * xv.x; ay += a * xv.y;
    }
    atomicAdd(&out[b * HD + d], ax);
    atomicAdd(&out[b * HD + d + 1], ay);
}

// ---------------------------------------------------------------------------
extern "C" void kernel_launch(void* const* d_in, const int* in_sizes, int n_in,
                              void* d_out, int out_size, void* d_ws, size_t ws_size,
                              hipStream_t stream)
{
    const float* x          = (const float*)d_in[0];
    const float* weight     = (const float*)d_in[1];
    const float* mask       = (const float*)d_in[2];
    const float* squish_w   = (const float*)d_in[3];
    const float* atten_proj = (const float*)d_in[4];
    float* out = (float*)d_out;

    float*    scores = (float*)d_ws;                              // 512 KB
    unsigned* Bf     = (unsigned*)((char*)d_ws + (size_t)BATCH * SEQ * 4); // 512 KB

    bprep_kernel<<<512, 256, 0, stream>>>(squish_w, Bf);

    dim3 g1((BATCH * SEQ) / ROWS_WG);     // 1024 workgroups
    att_scores_kernel<<<g1, 256, 0, stream>>>(weight, Bf, atten_proj, scores);

    att_softmax_kernel<<<BATCH, 256, 0, stream>>>(mask, scores);

    hipMemsetAsync(d_out, 0, (size_t)BATCH * HD * sizeof(float), stream);
    att_pool_kernel<<<dim3(SCHUNKS, BATCH), 256, 0, stream>>>(x, scores, out);
}
